// VectorQuantizer_3779571221171
// MI455X (gfx1250) — matrix operand order repaired and checked by reading, not compiled, not run on hardware
//
#include <hip/hip_runtime.h>

typedef __attribute__((ext_vector_type(16))) _Float16 v16h;
typedef __attribute__((ext_vector_type(8)))  _Float16 v8h;
typedef __attribute__((ext_vector_type(4)))  _Float16 v4h;
typedef __attribute__((ext_vector_type(8)))  float    v8f;
typedef __attribute__((ext_vector_type(4)))  float    v4f;
typedef __attribute__((ext_vector_type(4)))  int      v4i;

#define NUM_CODES   1024
#define DIM         256
#define N_ROWS      32768
#define ROWS_PER_WG 128           // 8 waves * 16 rows
#define CHUNK       256           // codes staged per LDS chunk
#define NCHUNK      (NUM_CODES / CHUNK)
#define ROWSTRIDE_H 296           // halfs per LDS code row (592B: 16B-aligned, bank-friendly pad)
#define B_BYTES     (CHUNK * ROWSTRIDE_H * 2)      // 151552
#define CN_OFF      B_BYTES                        // CHUNK floats = 1024B
#define IDX_OFF     (B_BYTES + CHUNK * 4)          // 128 ints = 512B
#define SMEM_BYTES  (IDX_OFF + ROWS_PER_WG * 4)
#define WS_EMBH_OFF 4096                           // f16 codebook in workspace after norms

#if __has_builtin(__builtin_amdgcn_global_load_async_to_lds_b128)
#define HAVE_ASYNC_LDS 1
typedef __attribute__((address_space(1))) v4i gas_v4i;   // global AS pointer-to-b128
typedef __attribute__((address_space(3))) v4i las_v4i;   // LDS AS pointer-to-b128
#endif

__device__ __forceinline__ void wait_async0() {
#if __has_builtin(__builtin_amdgcn_s_wait_asynccnt)
    __builtin_amdgcn_s_wait_asynccnt(0);
#else
    asm volatile("s_wait_asynccnt 0" ::: "memory");
#endif
}

// ---------- prelude: per-code squared norms + f16 codebook into workspace ----------
__global__ void vq_prep(const float* __restrict__ emb, float* __restrict__ cn,
                        _Float16* __restrict__ embh) {
    int k = blockIdx.x * blockDim.x + threadIdx.x;   // 1024 threads total
    const float* r = emb + (size_t)k * DIM;
    _Float16* o = embh + (size_t)k * DIM;
    float s = 0.f;
#pragma unroll
    for (int i = 0; i < DIM; i += 4) {
        v4f f = *(const v4f*)(r + i);
        s += f.x * f.x + f.y * f.y + f.z * f.z + f.w * f.w;
        v4h h;
        h[0] = (_Float16)f.x; h[1] = (_Float16)f.y;
        h[2] = (_Float16)f.z; h[3] = (_Float16)f.w;
        *(v4h*)(o + i) = h;
    }
    cn[k] = s;
}

// ---------- main fused kernel: distances (WMMA f16) + argmin + gather ----------
__global__ void __launch_bounds__(256)
vq_main(const float* __restrict__ z_e, const float* __restrict__ emb,
        const float* __restrict__ codeNorms, const _Float16* __restrict__ embh,
        float* __restrict__ zOut, float* __restrict__ zqOut) {
    extern __shared__ char smem[];
    _Float16* bsm   = (_Float16*)smem;            // staged f16 codebook chunk
    float*    cnsm  = (float*)(smem + CN_OFF);    // staged code norms
    int*      idxsm = (int*)(smem + IDX_OFF);     // per-WG winning indices

    const int tid   = threadIdx.x;
    const int lane  = tid & 31;
    const int wave  = tid >> 5;
    const int col   = lane & 15;                  // N-column within 16-wide tile
    const int hf    = lane >> 4;                  // half-wave select
    const int wgRow = blockIdx.x * ROWS_PER_WG;
    const int rowBase = wgRow + wave * 16;        // this wave's 16-row strip

    // ---- load this wave's A strip once (read-once: non-temporal), f32 -> f16 ----
    // A 16x32 f16: lane(row=col): elems 0..7  = K = ks*32 + hf*8 + e
    //                             elems 8..15 = K = ks*32 + 16 + hf*8 + e
    const float* zrow = z_e + (size_t)(rowBase + col) * DIM;
    v16h a[8];
#pragma unroll
    for (int ks = 0; ks < 8; ++ks) {
        const int k0 = ks * 32 + hf * 8;
        v4f f0 = __builtin_nontemporal_load((const v4f*)(zrow + k0));
        v4f f1 = __builtin_nontemporal_load((const v4f*)(zrow + k0 + 4));
        v4f f2 = __builtin_nontemporal_load((const v4f*)(zrow + k0 + 16));
        v4f f3 = __builtin_nontemporal_load((const v4f*)(zrow + k0 + 20));
        a[ks][0] = (_Float16)f0.x; a[ks][1] = (_Float16)f0.y;
        a[ks][2] = (_Float16)f0.z; a[ks][3] = (_Float16)f0.w;
        a[ks][4] = (_Float16)f1.x; a[ks][5] = (_Float16)f1.y;
        a[ks][6] = (_Float16)f1.z; a[ks][7] = (_Float16)f1.w;
        a[ks][8] = (_Float16)f2.x; a[ks][9] = (_Float16)f2.y;
        a[ks][10] = (_Float16)f2.z; a[ks][11] = (_Float16)f2.w;
        a[ks][12] = (_Float16)f3.x; a[ks][13] = (_Float16)f3.y;
        a[ks][14] = (_Float16)f3.z; a[ks][15] = (_Float16)f3.w;
    }

    float bestVal[8];
    int   bestIdx[8];
#pragma unroll
    for (int v = 0; v < 8; ++v) { bestVal[v] = 3.4e38f; bestIdx[v] = 0; }

    for (int chunk = 0; chunk < NCHUNK; ++chunk) {
        const int cbase = chunk * CHUNK;
        __syncthreads();
        // ---- stage 256-code f16 chunk into LDS (padded rows), ASYNCcnt path ----
        const _Float16* esrc = embh + (size_t)cbase * DIM;
#ifdef HAVE_ASYNC_LDS
#pragma unroll 4
        for (int i = 0; i < (CHUNK * DIM / 8) / 256; ++i) {   // 32 asyncs/thread
            int linear = i * 256 + tid;
            int code = linear >> 5;                  // 32 b128 per code row
            int q = linear & 31;
            __builtin_amdgcn_global_load_async_to_lds_b128(
                (gas_v4i*)(esrc + (size_t)code * DIM + q * 8),
                (las_v4i*)(bsm + code * ROWSTRIDE_H + q * 8), 0, 0);
        }
#else
#pragma unroll 4
        for (int i = 0; i < (CHUNK * DIM / 8) / 256; ++i) {
            int linear = i * 256 + tid;
            int code = linear >> 5;
            int q = linear & 31;
            v8h h = *(const v8h*)(esrc + (size_t)code * DIM + q * 8);
            *(v8h*)(bsm + code * ROWSTRIDE_H + q * 8) = h;
        }
#endif
        if (tid < CHUNK) cnsm[tid] = codeNorms[cbase + tid];
#ifdef HAVE_ASYNC_LDS
        wait_async0();
#endif
        __syncthreads();

        // ---- 16 code-tiles of 16 codes; force-preload B, then 8 b2b WMMAs ----
        for (int tile = 0; tile < CHUNK / 16; ++tile) {
            const int code = cbase + tile * 16 + col;
            const float cn = cnsm[tile * 16 + col];
            // B 32x16 f16: lane(col): elems 0..7 = K = ks*32 + hf*8 + e, elems 8..15 = K = ks*32 + 16 + hf*8 + e
            const _Float16* bp0 = bsm + (tile * 16 + col) * ROWSTRIDE_H + hf * 8;
            v16h b[8];
#pragma unroll
            for (int ks = 0; ks < 8; ++ks) {
                const _Float16* bp = bp0 + ks * 32;
                v8h lo = *(const v8h*)bp;
                v8h hi = *(const v8h*)(bp + 16);
                b[ks] = __builtin_shufflevector(lo, hi, 0, 1, 2, 3, 4, 5, 6, 7,
                                                8, 9, 10, 11, 12, 13, 14, 15);
            }
            // Keep all 8 B operands live so the 16 ds_loads issue under ONE
            // s_wait_dscnt and the 8 WMMAs go back-to-back.
            asm volatile("" : "+v"(b[0]), "+v"(b[1]), "+v"(b[2]), "+v"(b[3]),
                              "+v"(b[4]), "+v"(b[5]), "+v"(b[6]), "+v"(b[7]));
            v8f acc = {};
#pragma unroll
            for (int ks = 0; ks < 8; ++ks) {
                acc = __builtin_amdgcn_wmma_f32_16x16x32_f16(
                    false, a[ks], false, b[ks], (short)0, acc, false, false);
            }
#pragma unroll
            for (int v = 0; v < 8; ++v) {
                float s = cn - 2.0f * acc[v];        // ||x||^2 constant per row: dropped
                if (s < bestVal[v]) { bestVal[v] = s; bestIdx[v] = code; }
            }
        }
    }

    // ---- reduce over the 16-lane N dimension per output row ----
#pragma unroll
    for (int v = 0; v < 8; ++v) {
        float bv = bestVal[v];
        int   bi = bestIdx[v];
#pragma unroll
        for (int m = 8; m >= 1; m >>= 1) {
            float ov = __shfl_xor(bv, m, 32);
            int   oi = __shfl_xor(bi, m, 32);
            if (ov < bv || (ov == bv && oi < bi)) { bv = ov; bi = oi; }
        }
        if (col == 0) {
            int lrow = wave * 16 + v + 8 * hf;       // row = v + 8*half of strip
            idxsm[lrow] = bi;
            zOut[wgRow + lrow] = (float)bi;          // argmin index (float-encoded)
        }
    }
    __syncthreads();

    // ---- gather z_q = embeddings[z] : coalesced float4 copy (NT stores) ----
#pragma unroll 4
    for (int i = 0; i < (ROWS_PER_WG * DIM / 4) / 256; ++i) {  // 32 iters
        int linear = i * 256 + tid;
        int lrow = linear >> 6;
        int q = linear & 63;
        int code = idxsm[lrow];
        v4f f = *(const v4f*)(emb + (size_t)code * DIM + q * 4);
        __builtin_nontemporal_store(f, (v4f*)(zqOut + (size_t)(wgRow + lrow) * DIM + q * 4));
    }
}

extern "C" void kernel_launch(void* const* d_in, const int* in_sizes, int n_in,
                              void* d_out, int out_size, void* d_ws, size_t ws_size,
                              hipStream_t stream) {
    const float* z_e = (const float*)d_in[0];
    const float* emb = (const float*)d_in[1];
    float*     cn   = (float*)d_ws;                           // 1024 floats
    _Float16*  embh = (_Float16*)((char*)d_ws + WS_EMBH_OFF); // 1024*256 halfs
    float* zOut = (float*)d_out;                              // 32768 indices (as float)
    float* zq   = zOut + N_ROWS;                              // 32768*256 floats

    (void)in_sizes; (void)n_in; (void)out_size; (void)ws_size;

    (void)hipFuncSetAttribute((const void*)vq_main,
                              hipFuncAttributeMaxDynamicSharedMemorySize, SMEM_BYTES);

    vq_prep<<<NUM_CODES / 256, 256, 0, stream>>>(emb, cn, embh);
    vq_main<<<N_ROWS / ROWS_PER_WG, 256, SMEM_BYTES, stream>>>(z_e, emb, cn, embh, zOut, zq);
}
